// LinearCrossAttention_80985903333479
// MI455X (gfx1250) — matrix operand order repaired and checked by reading, not compiled, not run on hardware
//
#include <hip/hip_runtime.h>
#include <hip/hip_bf16.h>

typedef __attribute__((ext_vector_type(16))) _Float16 v16h;
typedef __attribute__((ext_vector_type(8)))  float    v8f;
typedef __attribute__((ext_vector_type(2)))  __fp16   fp16v2;

#define CC 1024          // channels
#define HH 16            // heads
#define DD 64            // head dim
#define BM 128
#define BN 64
#define BK 32

struct alignas(8) h4 { _Float16 a, b, c, d; };

static __device__ __forceinline__ v8f wmma_f16(const v16h& a, const v16h& b, const v8f& c) {
    return __builtin_amdgcn_wmma_f32_16x16x32_f16(false, a, false, b, (short)0, c, false, false);
}

// packed f32x2 -> f16x2 conversions (v_cvt_pkrtz_f16_f32)
static __device__ __forceinline__ h4 cvt4(const float4& f) {
    fp16v2 lo = __builtin_amdgcn_cvt_pkrtz(f.x, f.y);
    fp16v2 hi = __builtin_amdgcn_cvt_pkrtz(f.z, f.w);
    h4 r;
    r.a = (_Float16)lo[0]; r.b = (_Float16)lo[1];
    r.c = (_Float16)hi[0]; r.d = (_Float16)hi[1];
    return r;
}

// ---------------------------------------------------------------------------
// Y(M,N) = A(M,K) @ W(N,K)^T + bias ; A,W f32 -> f16 while staging into LDS.
// Block: 256 threads = 8 waves. Tile: BM=128 x BN=64, K step 32.
// ---------------------------------------------------------------------------
__global__ __launch_bounds__(256)
void lca_gemm_xwT(const float* __restrict__ A, const float* __restrict__ W,
                  const float* __restrict__ bias,
                  float* __restrict__ out32, _Float16* __restrict__ out16,
                  int N, int K)
{
    __shared__ _Float16 As[BM][BK + 8];   // row stride 40 halfs = 80 B (8B aligned)
    __shared__ _Float16 Ws[BN][BK + 8];

    const int tid  = threadIdx.x;
    const int lane = tid & 31;
    const int w    = tid >> 5;                 // wave id 0..7 -> row strip
    const long m0  = (long)blockIdx.y * BM;
    const int  n0  = blockIdx.x * BN;

    const v8f vzero = {0.f,0.f,0.f,0.f,0.f,0.f,0.f,0.f};
    v8f acc[4];
    #pragma unroll
    for (int c = 0; c < 4; ++c) acc[c] = vzero;

    for (int k0 = 0; k0 < K; k0 += BK) {
        if (k0 + BK < K) {   // prefetch next K-chunk (global_prefetch_b8)
            __builtin_prefetch(&A[(m0 + (tid >> 1)) * K + k0 + BK], 0, 3);
            __builtin_prefetch(&W[((long)(n0 + (tid >> 2))) * K + k0 + BK], 0, 3);
        }
        // ---- stage A: 128x32 f32 = 1024 float4 groups, 4 per thread ----
        float4 areg[4];
        #pragma unroll
        for (int j = 0; j < 4; ++j) {
            const int g = j * 256 + tid;          // group id
            const int r = g >> 3, c4 = g & 7;     // 8 float4 per row
            areg[j] = *(const float4*)&A[(m0 + r) * K + k0 + c4 * 4];
        }
        // ---- stage W: 64x32 f32 = 512 float4 groups, 2 per thread ----
        float4 wreg[2];
        #pragma unroll
        for (int j = 0; j < 2; ++j) {
            const int g = j * 256 + tid;
            const int r = g >> 3, c4 = g & 7;
            wreg[j] = *(const float4*)&W[((long)(n0 + r)) * K + k0 + c4 * 4];
        }
        #pragma unroll
        for (int j = 0; j < 4; ++j) {
            const int g = j * 256 + tid;
            const int r = g >> 3, c4 = g & 7;
            *(h4*)&As[r][c4 * 4] = cvt4(areg[j]);
        }
        #pragma unroll
        for (int j = 0; j < 2; ++j) {
            const int g = j * 256 + tid;
            const int r = g >> 3, c4 = g & 7;
            *(h4*)&Ws[r][c4 * 4] = cvt4(wreg[j]);
        }
        __syncthreads();

        // Preload A fragment + all 4 B fragments, then issue 4 WMMAs back-to-back
        v16h a;
        {
            const int mrow = w * 16 + (lane & 15);
            const int kb   = (lane < 16) ? 0 : 8;
            #pragma unroll
            for (int i = 0; i < 8; ++i) {
                a[i]     = As[mrow][kb + i];
                a[8 + i] = As[mrow][kb + 16 + i];
            }
        }
        v16h bf[4];
        {
            const int ko = (lane < 16) ? 0 : 8;
            #pragma unroll
            for (int c = 0; c < 4; ++c) {
                const int n = c * 16 + (lane & 15);
                #pragma unroll
                for (int i = 0; i < 16; ++i) bf[c][i] = Ws[n][ko + i + (i & 8)];
            }
        }
        #pragma unroll
        for (int c = 0; c < 4; ++c) acc[c] = wmma_f16(a, bf[c], acc[c]);
        __syncthreads();
    }

    // Epilogue: C/D layout -> VGPR r: lanes0-15 M=r, lanes16-31 M=r+8; N=lane&15
    #pragma unroll
    for (int c = 0; c < 4; ++c) {
        const int col = n0 + c * 16 + (lane & 15);
        const float bb = bias[col];
        #pragma unroll
        for (int r = 0; r < 8; ++r) {
            const long row = m0 + w * 16 + r + ((lane < 16) ? 0 : 8);
            const float val = acc[c][r] + bb;
            if (out32) out32[row * N + col] = val;
            if (out16) out16[row * N + col] = (_Float16)val;
        }
    }
}

// ---------------------------------------------------------------------------
// Softmax over contiguous 64-wide head slices. One wave per group.
// ---------------------------------------------------------------------------
__global__ __launch_bounds__(256)
void lca_softmax64(const float* __restrict__ in, float* __restrict__ out32,
                   _Float16* __restrict__ out16, long nGroups)
{
    const long g  = ((long)blockIdx.x * blockDim.x + threadIdx.x) >> 5;
    const int lane = threadIdx.x & 31;
    if (g >= nGroups) return;
    const float* p = in + g * 64;
    float a0 = p[lane], a1 = p[lane + 32];
    float m = fmaxf(a0, a1);
    #pragma unroll
    for (int off = 16; off; off >>= 1) m = fmaxf(m, __shfl_xor(m, off, 32));
    float e0 = __expf(a0 - m), e1 = __expf(a1 - m);
    float s = e0 + e1;
    #pragma unroll
    for (int off = 16; off; off >>= 1) s += __shfl_xor(s, off, 32);
    const float inv = 1.0f / s;
    e0 *= inv; e1 *= inv;
    if (out32) { out32[g * 64 + lane] = e0; out32[g * 64 + lane + 32] = e1; }
    out16[g * 64 + lane]      = (_Float16)e0;
    out16[g * 64 + lane + 32] = (_Float16)e1;
}

// ---------------------------------------------------------------------------
// ksum[b*CC + c] = sum_t k[b,t,c]. Grid = B*CC blocks.
// ---------------------------------------------------------------------------
__global__ __launch_bounds__(256)
void lca_colsum(const _Float16* __restrict__ hk, float* __restrict__ ksum, int T)
{
    const int b   = blockIdx.x >> 10;
    const int col = blockIdx.x & (CC - 1);
    const long base = ((long)b * T) * CC + col;
    float s = 0.f;
    for (int t = threadIdx.x; t < T; t += 256) s += (float)hk[base + (long)t * CC];
    #pragma unroll
    for (int off = 16; off; off >>= 1) s += __shfl_xor(s, off, 32);
    __shared__ float red[8];
    if ((threadIdx.x & 31) == 0) red[threadIdx.x >> 5] = s;
    __syncthreads();
    if (threadIdx.x == 0) {
        float t = 0.f;
        #pragma unroll
        for (int i = 0; i < 8; ++i) t += red[i];
        ksum[blockIdx.x] = t;
    }
}

// ---------------------------------------------------------------------------
// kv[b,h] = k^T @ v : (64 x T) @ (T x 64). One block per (b,h); 8 waves share
// each K-chunk; wave w owns row-tile w&3 and col-tiles 2*(w>>2), 2*(w>>2)+1.
// k is staged TRANSPOSED into LDS so A-fragment reads are contiguous b128.
// ---------------------------------------------------------------------------
__global__ __launch_bounds__(256)
void lca_kv_gemm(const _Float16* __restrict__ hk, const _Float16* __restrict__ hv,
                 _Float16* __restrict__ hkv, int T)
{
    __shared__ _Float16 kst[DD][BK + 8];   // k^T: [d][t], row stride 40 halfs
    __shared__ _Float16 vs[32][72];        // v:   [t][e]
    const int bh = blockIdx.x;
    const int b = bh >> 4, h = bh & 15;
    const int tid = threadIdx.x, lane = tid & 31, w = tid >> 5;
    const int rt  = w & 3;
    const int ct0 = (w >> 2) * 2;
    const v8f vzero = {0.f,0.f,0.f,0.f,0.f,0.f,0.f,0.f};
    v8f acc0 = vzero, acc1 = vzero;
    const long base = ((long)b * T) * CC + h * DD;

    for (int t0 = 0; t0 < T; t0 += 32) {
        // 32x64 halfs per array = 512 h4 groups, 2 per thread
        h4 kreg[2], vreg[2];
        #pragma unroll
        for (int j = 0; j < 2; ++j) {
            const int g = j * 256 + tid;
            const int r = g >> 4, c4 = g & 15;       // 16 h4 groups per row of 64
            const long gofs = base + (long)(t0 + r) * CC + c4 * 4;
            kreg[j] = *(const h4*)&hk[gofs];
            vreg[j] = *(const h4*)&hv[gofs];
        }
        #pragma unroll
        for (int j = 0; j < 2; ++j) {
            const int g = j * 256 + tid;
            const int r = g >> 4, c4 = g & 15;
            // k transposed scatter: kst[d][t]
            kst[c4 * 4 + 0][r] = kreg[j].a;
            kst[c4 * 4 + 1][r] = kreg[j].b;
            kst[c4 * 4 + 2][r] = kreg[j].c;
            kst[c4 * 4 + 3][r] = kreg[j].d;
            *(h4*)&vs[r][c4 * 4] = vreg[j];
        }
        __syncthreads();
        // A = k^T, now row-major in LDS: contiguous fragment reads
        v16h a;
        {
            const int d  = rt * 16 + (lane & 15);
            const int kb = (lane < 16) ? 0 : 8;
            #pragma unroll
            for (int i = 0; i < 8; ++i) { a[i] = kst[d][kb + i]; a[8 + i] = kst[d][kb + 16 + i]; }
        }
        v16h b0, b1;
        {
            const int ko = (lane < 16) ? 0 : 8;
            const int e0 = ct0 * 16 + (lane & 15);
            #pragma unroll
            for (int i = 0; i < 16; ++i) { b0[i] = vs[ko + i + (i & 8)][e0]; b1[i] = vs[ko + i + (i & 8)][e0 + 16]; }
        }
        acc0 = wmma_f16(a, b0, acc0);
        acc1 = wmma_f16(a, b1, acc1);
        __syncthreads();
    }
    #pragma unroll
    for (int r = 0; r < 8; ++r) {
        const int drow = rt * 16 + r + ((lane < 16) ? 0 : 8);
        const int e    = ct0 * 16 + (lane & 15);
        const long o   = ((long)bh * 64 + drow) * 64;
        hkv[o + e]      = (_Float16)acc0[r];
        hkv[o + e + 16] = (_Float16)acc1[r];
    }
}

// ---------------------------------------------------------------------------
// attn[m, h*64+e] += (q[m,h,:] @ kv[b,h]) * Dinv(m,h), Dinv = 1/max(q.ksum,eps)
// Grid: (M/128, H). Block 256 = 8 waves, each wave 16 rows x 64 cols.
// ---------------------------------------------------------------------------
__global__ __launch_bounds__(256)
void lca_qkv_acc(const _Float16* __restrict__ hq, const _Float16* __restrict__ hkv,
                 const float* __restrict__ ksum, float* __restrict__ attn, int T)
{
    __shared__ _Float16 qs[128][72];
    __shared__ _Float16 kvs[64][72];
    __shared__ float dinv[128];
    const int tid = threadIdx.x, lane = tid & 31, w = tid >> 5;
    const long m0 = (long)blockIdx.x * 128;
    const int h  = blockIdx.y;
    const int b  = (int)(m0 / T);
    const int bh = b * HH + h;

    // stage kv 64x64 = 1024 h4 groups (4/thread), q 128x64 = 2048 h4 (8/thread)
    h4 kvreg[4], qreg[8];
    #pragma unroll
    for (int j = 0; j < 4; ++j) {
        const int g = j * 256 + tid;
        const int r = g >> 4, c4 = g & 15;
        kvreg[j] = *(const h4*)&hkv[((long)bh * 64 + r) * 64 + c4 * 4];
    }
    #pragma unroll
    for (int j = 0; j < 8; ++j) {
        const int g = j * 256 + tid;
        const int r = g >> 4, c4 = g & 15;
        qreg[j] = *(const h4*)&hq[(m0 + r) * CC + h * DD + c4 * 4];
    }
    #pragma unroll
    for (int j = 0; j < 4; ++j) {
        const int g = j * 256 + tid;
        const int r = g >> 4, c4 = g & 15;
        *(h4*)&kvs[r][c4 * 4] = kvreg[j];
    }
    #pragma unroll
    for (int j = 0; j < 8; ++j) {
        const int g = j * 256 + tid;
        const int r = g >> 4, c4 = g & 15;
        *(h4*)&qs[r][c4 * 4] = qreg[j];
    }
    __syncthreads();
    if (tid < 128) {
        float s = 0.f;
        const float* kp = ksum + (long)bh * DD;
        #pragma unroll
        for (int d = 0; d < DD; ++d) s += (float)qs[tid][d] * kp[d];
        dinv[tid] = 1.0f / fmaxf(s, 1e-9f);
    }
    __syncthreads();

    const v8f vzero = {0.f,0.f,0.f,0.f,0.f,0.f,0.f,0.f};
    v8f acc[4];
    #pragma unroll
    for (int c = 0; c < 4; ++c) acc[c] = vzero;

    #pragma unroll
    for (int k0 = 0; k0 < 64; k0 += 32) {
        v16h a;
        const int mrow = w * 16 + (lane & 15);
        const int kb   = k0 + ((lane < 16) ? 0 : 8);
        #pragma unroll
        for (int i = 0; i < 8; ++i) { a[i] = qs[mrow][kb + i]; a[8 + i] = qs[mrow][kb + 16 + i]; }
        v16h bf[4];
        {
            const int ko = k0 + ((lane < 16) ? 0 : 8);
            #pragma unroll
            for (int ct = 0; ct < 4; ++ct) {
                const int n = ct * 16 + (lane & 15);
                #pragma unroll
                for (int i = 0; i < 16; ++i) bf[ct][i] = kvs[ko + i + (i & 8)][n];
            }
        }
        #pragma unroll
        for (int ct = 0; ct < 4; ++ct) acc[ct] = wmma_f16(a, bf[ct], acc[ct]);
    }
    #pragma unroll
    for (int ct = 0; ct < 4; ++ct) {
        #pragma unroll
        for (int r = 0; r < 8; ++r) {
            const int rl  = w * 16 + r + ((lane < 16) ? 0 : 8);
            const long row = m0 + rl;
            const int col = h * DD + ct * 16 + (lane & 15);
            attn[row * CC + col] += acc[ct][r] * dinv[rl];
        }
    }
}

// ---------------------------------------------------------------------------
extern "C" void kernel_launch(void* const* d_in, const int* in_sizes, int n_in,
                              void* d_out, int out_size, void* d_ws, size_t ws_size,
                              hipStream_t stream)
{
    (void)in_sizes; (void)n_in; (void)out_size; (void)ws_size;
    const float* x  = (const float*)d_in[0];
    const float* y0 = (const float*)d_in[1];
    const float* y1 = (const float*)d_in[2];
    const float* Wq = (const float*)d_in[3];
    const float* bq = (const float*)d_in[4];
    const float* Wk = (const float*)d_in[5];
    const float* bk = (const float*)d_in[6];
    const float* Wv = (const float*)d_in[7];
    const float* bv = (const float*)d_in[8];
    const float* Wp = (const float*)d_in[9];
    const float* bp = (const float*)d_in[10];
    float* out = (float*)d_out;

    const int  B = 4, TQ = 4096, TB = 4096;
    const long M = (long)B * TQ;                 // 16384
    const long elems = M * CC;                   // 16.78M

    char* ws = (char*)d_ws;
    size_t off = 0;
    float*    proj32 = (float*)(ws + off);    off += elems * 4;           // 64 MB
    float*    attn32 = (float*)(ws + off);    off += elems * 4;           // 64 MB
    _Float16* hq     = (_Float16*)(ws + off); off += elems * 2;           // 32 MB
    _Float16* hk     = (_Float16*)(ws + off); off += elems * 2;           // 32 MB
    _Float16* hv     = (_Float16*)(ws + off); off += elems * 2;           // 32 MB
    _Float16* hkv    = (_Float16*)(ws + off); off += (size_t)B * HH * DD * DD * 2;
    float*    ksum   = (float*)(ws + off);    off += (size_t)B * CC * 4;

    const dim3 gGemm(CC / BN, (unsigned)(M / BM));   // (16, 128)
    const long nGroups = elems / 64;                 // 262144
    const int  gSoft = (int)(nGroups * 32 / 256);    // 32768 blocks

    // q projection + softmax (writes attn32 init and hq)
    lca_gemm_xwT<<<gGemm, 256, 0, stream>>>(x, Wq, bq, proj32, nullptr, CC, CC);
    lca_softmax64<<<gSoft, 256, 0, stream>>>(proj32, attn32, hq, nGroups);

    for (int i = 0; i < 2; ++i) {
        const float* y = (i == 0) ? y0 : y1;
        const float* Wki = Wk + (size_t)i * CC * CC;
        const float* bki = bk + (size_t)i * CC;
        const float* Wvi = Wv + (size_t)i * CC * CC;
        const float* bvi = bv + (size_t)i * CC;

        lca_gemm_xwT<<<gGemm, 256, 0, stream>>>(y, Wki, bki, proj32, nullptr, CC, CC);
        lca_softmax64<<<gSoft, 256, 0, stream>>>(proj32, nullptr, hk, nGroups);
        lca_colsum<<<B * CC, 256, 0, stream>>>(hk, ksum, TB);
        lca_gemm_xwT<<<gGemm, 256, 0, stream>>>(y, Wvi, bvi, nullptr, hv, CC, CC);
        lca_kv_gemm<<<B * HH, 256, 0, stream>>>(hk, hv, hkv, TB);
        lca_qkv_acc<<<dim3((unsigned)(M / 128), HH), 256, 0, stream>>>(hq, hkv, ksum, attn32, TQ);
    }

    // final projection into d_out
    lca_gemm_xwT<<<gGemm, 256, 0, stream>>>(attn32, Wp, bp, out, nullptr, CC, CC);
}
